// ExpansionContrastModule_39994735460364
// MI455X (gfx1250) — matrix operand order repaired and checked by reading, not compiled, not run on hardware
//
#include <hip/hip_runtime.h>
#include <math.h>

// ---------------------------------------------------------------------------
// Types / helpers for CDNA5 WMMA (wave32, v_wmma_f32_16x16x32_bf16)
// ---------------------------------------------------------------------------
typedef __attribute__((ext_vector_type(16))) __bf16 v16bf;
typedef __attribute__((ext_vector_type(8)))  float  v8f;

union Frag16 { v16bf v; unsigned int u[8]; };

__device__ __forceinline__ unsigned short f2bf(float f) {
  union { float f; unsigned int u; } x; x.f = f;
  unsigned int r = x.u + 0x7FFFu + ((x.u >> 16) & 1u);  // RNE
  return (unsigned short)(r >> 16);
}
__device__ __forceinline__ float bf2f(unsigned short h) {
  union { float f; unsigned int u; } x; x.u = ((unsigned int)h) << 16; return x.f;
}
__device__ __forceinline__ v8f wmma_bf16(const Frag16& a, const Frag16& b, v8f c) {
  // D = A(16x32 bf16) * B(32x16 bf16) + C(16x16 f32)
  return __builtin_amdgcn_wmma_f32_16x16x32_bf16(false, a.v, false, b.v,
                                                 (short)0, c, false, false);
}

// Problem constants
#define BB   8
#define CC   64
#define HH   96
#define WW   96
#define HWW  9216     // 96*96
#define NLK  8        // num fixed contrast kernels
#define NHD  4        // heads
#define HID  16       // C / NH
#define KDIM 512      // NL*C
#define MDIM 128      // NL*HID
#define NTILES 576    // HW/16

// Fixed base kernels (row-major 3x3) and their mean
__constant__ float BASEK[8][9] = {
  {-1,0,0,  0,1,0,  0,0,0},
  { 0,-1,0, 0,1,0,  0,0,0},
  { 0,0,-1, 0,1,0,  0,0,0},
  { 0,0,0,  0,1,-1, 0,0,0},
  { 0,0,0,  0,1,0,  0,0,-1},
  { 0,0,0,  0,1,0,  0,-1,0},
  { 0,0,0,  0,1,0, -1,0,0},
  { 0,0,0, -1,1,0,  0,0,0}};
__constant__ float K0MEAN[9] =
  {-0.125f,-0.125f,-0.125f, -0.125f,1.0f,-0.125f, -0.125f,-0.125f,-0.125f};

// Workspace layout (bytes, 256-aligned)
#define OFF_MIXED   ((size_t)0)            //  4*8*64*9 f32  = 73728
#define OFF_KWB     ((size_t)73728)        //  4*128*512 bf16 = 524288
#define OFF_VWB     ((size_t)598016)
#define OFF_QWB     ((size_t)1122304)      //  4*16*64 bf16 = 8192
#define OFF_OUTWB   ((size_t)1130496)      //  64*64 bf16   = 8192
#define OFF_KEYSB   ((size_t)1138688)      //  8*4*128*9216 bf16 = 75497472
#define OFF_VALSB   ((size_t)76636160)
#define OFF_QRYB    ((size_t)152133632)    //  8*4*16*9216 bf16 = 9437184
#define OFF_INVNK   ((size_t)161570816)    //  4096 f32 = 16384
#define OFF_INVNQ   ((size_t)161587200)    //  512 f32  = 2048
#define OFF_SCORES  ((size_t)161589248)    //  32*2048 f32 = 262144
#define OFF_ATTNB   ((size_t)161851392)    //  32*2048 bf16 = 131072
#define OFF_OUTB    ((size_t)161982464)    //  8*64*9216 bf16 = 9437184
#define OFF_YBUF    ((size_t)171419648)    //  8*64*9216 f32 = 18874368
#define OFF_CSUM    ((size_t)190294016)    //  64 f32
#define OFF_CSSQ    ((size_t)190294272)
#define WS_NEEDED   ((size_t)190294528)

// ---------------------------------------------------------------------------
// K1: softmax(sum_weights) -> mixed kernels; convert weights to bf16; zero BN
// ---------------------------------------------------------------------------
__global__ __launch_bounds__(256) void ecm_prep(
    const float* __restrict__ sumw, const float* __restrict__ qw,
    const float* __restrict__ kw,   const float* __restrict__ vw,
    const float* __restrict__ outw,
    float* __restrict__ mixedG, unsigned short* __restrict__ qwB,
    unsigned short* __restrict__ kwB, unsigned short* __restrict__ vwB,
    unsigned short* __restrict__ outwB,
    float* __restrict__ chanSum, float* __restrict__ chanSumSq) {
  int tid = blockIdx.x * 256 + threadIdx.x;
  if (tid < 262144) { kwB[tid] = f2bf(kw[tid]); vwB[tid] = f2bf(vw[tid]); }
  if (tid < 4096)   { qwB[tid] = f2bf(qw[tid]); outwB[tid] = f2bf(outw[tid]); }
  if (tid < 64)     { chanSum[tid] = 0.f; chanSumSq[tid] = 0.f; }
  if (tid < 18432) {   // layout [i][j][c][t], t fastest
    int t = tid % 9;
    int c = (tid / 9) % 64;
    int j = (tid / (9 * 64)) % 8;
    int i = tid / (9 * 64 * 8);
    float e0 = sumw[(i * 64 + c) * 2 + 0];
    float e1 = sumw[(i * 64 + c) * 2 + 1];
    float m  = fmaxf(e0, e1);
    float s0 = __expf(e0 - m), s1 = __expf(e1 - m);
    float inv = 1.f / (s0 + s1);
    mixedG[tid] = BASEK[j][t] * (s0 * inv) + K0MEAN[t] * (s1 * inv);
  }
}

// ---------------------------------------------------------------------------
// K2: fused dilated depthwise conv (surround) -> LDS bf16 B-tile ->
//     WMMA projections keys = k_w @ surr, vals = v_w @ surr
// grid (576 tiles, 8 batch, 4 shifts), block 256 = 8 waves
// ---------------------------------------------------------------------------
__global__ __launch_bounds__(256) void ecm_conv_proj(
    const float* __restrict__ cen, const float* __restrict__ mixedG,
    const unsigned short* __restrict__ kwB, const unsigned short* __restrict__ vwB,
    unsigned short* __restrict__ keysB, unsigned short* __restrict__ valsB) {
  const int stile = blockIdx.x, b = blockIdx.y, ish = blockIdx.z;
  const int dil = 1 << ish;                 // shifts 1,2,4,8
  const int tid = threadIdx.x;
  __shared__ float          wmix[NLK * CC * 9];   // 18 KB
  __shared__ unsigned short Btile[16][KDIM];      // 16 KB  [n][k]
  for (int idx = tid; idx < NLK * CC * 9; idx += 256)
    wmix[idx] = mixedG[ish * (NLK * CC * 9) + idx];
  __syncthreads();

  const int sbase = stile * 16;
  for (int pair = tid; pair < CC * 16; pair += 256) {
    int c = pair >> 4, p = pair & 15;
    int pix = sbase + p;
    int y = pix / WW, x = pix - y * WW;
    const float* cptr = cen + (size_t)(b * CC + c) * HWW;
    float tap[9];
#pragma unroll
    for (int r = 0; r < 3; ++r)
#pragma unroll
      for (int cc2 = 0; cc2 < 3; ++cc2) {
        int yy = y + (r - 1) * dil, xx = x + (cc2 - 1) * dil;
        tap[r * 3 + cc2] = (yy >= 0 && yy < HH && xx >= 0 && xx < WW)
                               ? cptr[yy * WW + xx] : 0.f;
      }
#pragma unroll
    for (int j = 0; j < NLK; ++j) {
      const float* wp = &wmix[(j * CC + c) * 9];
      float acc = 0.f;
#pragma unroll
      for (int t = 0; t < 9; ++t) acc = fmaf(wp[t], tap[t], acc);
      Btile[p][j * CC + c] = f2bf(acc);     // cat order: channel = j*64+c
    }
  }
  __syncthreads();

  const int wave = tid >> 5, lane = tid & 31, half = lane >> 4, l15 = lane & 15;
  const unsigned short* kwRow = kwB + (size_t)(ish * MDIM + wave * 16 + l15) * KDIM;
  const unsigned short* vwRow = vwB + (size_t)(ish * MDIM + wave * 16 + l15) * KDIM;
  v8f accK = {}; v8f accV = {};
  for (int k0 = 0; k0 < KDIM; k0 += 32) {
    Frag16 a_k, a_v, bf;
#pragma unroll
    for (int r = 0; r < 8; ++r) {
      int ka = k0 + half * 16 + 2 * r;
      a_k.u[r] = *(const unsigned int*)(kwRow + ka);
      a_v.u[r] = *(const unsigned int*)(vwRow + ka);
      int kb = k0 + half * 16 + 2 * r;
      bf.u[r]  = *(const unsigned int*)(&Btile[l15][kb]);
    }
    accK = wmma_bf16(a_k, bf, accK);
    accV = wmma_bf16(a_v, bf, accV);
  }
#pragma unroll
  for (int r = 0; r < 8; ++r) {
    int ch  = wave * 16 + r + 8 * half;     // D row
    int lin = ch * 4 + ish;                 // stack(dim=2).view reinterpret
    int head = lin >> 7, q = lin & 127;
    size_t off = ((size_t)((b * NHD + head) * MDIM + q)) * HWW + sbase + l15;
    keysB[off] = f2bf(accK[r]);
    valsB[off] = f2bf(accV[r]);
  }
}

// ---------------------------------------------------------------------------
// K3: qrys = q_w @ cen  (M=16, K=64) via WMMA, one wave per tile
// ---------------------------------------------------------------------------
__global__ __launch_bounds__(32) void ecm_qry(
    const float* __restrict__ cen, const unsigned short* __restrict__ qwB,
    unsigned short* __restrict__ qryB) {
  const int stile = blockIdx.x, b = blockIdx.y, ish = blockIdx.z;
  const int lane = threadIdx.x, half = lane >> 4, l15 = lane & 15;
  const int sbase = stile * 16;
  __shared__ unsigned short Bt[16][CC];
  for (int idx = lane; idx < CC * 16; idx += 32) {
    int c = idx >> 4, p = idx & 15;
    Bt[p][c] = f2bf(cen[(size_t)(b * CC + c) * HWW + sbase + p]);
  }
  __syncthreads();
  const unsigned short* arow = qwB + (size_t)(ish * HID + l15) * CC;
  v8f acc = {};
#pragma unroll
  for (int k0 = 0; k0 < CC; k0 += 32) {
    Frag16 a, bf;
#pragma unroll
    for (int r = 0; r < 8; ++r) {
      int ka = k0 + half * 16 + 2 * r;
      a.u[r] = *(const unsigned int*)(arow + ka);
      int kb = k0 + half * 16 + 2 * r;
      bf.u[r] = *(const unsigned int*)(&Bt[l15][kb]);
    }
    acc = wmma_bf16(a, bf, acc);
  }
#pragma unroll
  for (int r = 0; r < 8; ++r) {
    int ch = r + 8 * half;
    int lin = ch * 4 + ish;
    int head = lin >> 4, q = lin & 15;
    qryB[((size_t)((b * NHD + head) * HID + q)) * HWW + sbase + l15] = f2bf(acc[r]);
  }
}

// ---------------------------------------------------------------------------
// K4: per-row inverse L2 norm for keys (4096 rows) and qrys (512 rows)
// ---------------------------------------------------------------------------
__global__ __launch_bounds__(256) void ecm_rownorm(
    const unsigned short* __restrict__ keysB, const unsigned short* __restrict__ qryB,
    float* __restrict__ invnk, float* __restrict__ invnq) {
  int row = blockIdx.x, tid = threadIdx.x;
  const unsigned short* p = (row < 4096) ? (keysB + (size_t)row * HWW)
                                         : (qryB + (size_t)(row - 4096) * HWW);
  float s = 0.f;
  for (int i = tid; i < HWW; i += 256) { float v = bf2f(p[i]); s = fmaf(v, v, s); }
  __shared__ float red[256];
  red[tid] = s; __syncthreads();
  for (int st = 128; st > 0; st >>= 1) {
    if (tid < st) red[tid] += red[tid + st];
    __syncthreads();
  }
  if (tid == 0) {
    float inv = 1.f / fmaxf(sqrtf(red[0]), 1e-12f);
    if (row < 4096) invnk[row] = inv; else invnq[row - 4096] = inv;
  }
}

// ---------------------------------------------------------------------------
// K5: scores[bh] = (q_n @ k_n^T)/96 : M=16, N=128, K=9216 via WMMA
// grid 32 (bh), block 256: wave = N-tile
// ---------------------------------------------------------------------------
__global__ __launch_bounds__(256) void ecm_scores(
    const unsigned short* __restrict__ qryB, const unsigned short* __restrict__ keysB,
    const float* __restrict__ invnq, const float* __restrict__ invnk,
    float* __restrict__ scores) {
  const int bh = blockIdx.x;
  const int wave = threadIdx.x >> 5, lane = threadIdx.x & 31;
  const int half = lane >> 4, l15 = lane & 15;
  const unsigned short* arow = qryB + (size_t)(bh * HID + l15) * HWW;
  const unsigned short* brow = keysB + (size_t)(bh * MDIM + wave * 16 + l15) * HWW;
  v8f acc = {};
  for (int k0 = 0; k0 < HWW; k0 += 32) {
    Frag16 a, bf;
#pragma unroll
    for (int r = 0; r < 8; ++r) {
      int ka = k0 + half * 16 + 2 * r;
      a.u[r] = *(const unsigned int*)(arow + ka);
      int kb = k0 + half * 16 + 2 * r;
      bf.u[r] = *(const unsigned int*)(brow + kb);
    }
    acc = wmma_bf16(a, bf, acc);
  }
#pragma unroll
  for (int r = 0; r < 8; ++r) {
    int q = r + 8 * half;
    int n = wave * 16 + l15;
    float v = acc[r] * invnq[bh * HID + q] * invnk[bh * MDIM + n] * (1.0f / 96.0f);
    scores[(size_t)bh * 2048 + q * 128 + n] = v;
  }
}

// ---------------------------------------------------------------------------
// K6: instance-norm over (q,k) + softmax over k -> bf16 attn
// ---------------------------------------------------------------------------
__global__ __launch_bounds__(256) void ecm_attn(
    const float* __restrict__ scores, unsigned short* __restrict__ attnB) {
  const int bh = blockIdx.x, tid = threadIdx.x;
  __shared__ float sc[2048];
  __shared__ float red[256], red2[256], rbuf[256], rbuf2[256];
  float s1 = 0.f, s2 = 0.f;
  for (int i = tid; i < 2048; i += 256) {
    float v = scores[(size_t)bh * 2048 + i];
    sc[i] = v; s1 += v; s2 = fmaf(v, v, s2);
  }
  red[tid] = s1; red2[tid] = s2; __syncthreads();
  for (int st = 128; st > 0; st >>= 1) {
    if (tid < st) { red[tid] += red[tid + st]; red2[tid] += red2[tid + st]; }
    __syncthreads();
  }
  float mean = red[0] * (1.f / 2048.f);
  float var  = red2[0] * (1.f / 2048.f) - mean * mean;
  float rs   = rsqrtf(var + 1e-5f);
  const int q = tid >> 4, g = tid & 15;
  float mx = -3.4e38f;
#pragma unroll
  for (int e = 0; e < 8; ++e)
    mx = fmaxf(mx, (sc[q * 128 + g * 8 + e] - mean) * rs);
  rbuf[tid] = mx; __syncthreads();
  for (int t = 0; t < 16; ++t) mx = fmaxf(mx, rbuf[q * 16 + t]);
  float ls = 0.f;
#pragma unroll
  for (int e = 0; e < 8; ++e)
    ls += __expf((sc[q * 128 + g * 8 + e] - mean) * rs - mx);
  rbuf2[tid] = ls; __syncthreads();
  float tot = 0.f;
  for (int t = 0; t < 16; ++t) tot += rbuf2[q * 16 + t];
  float inv = 1.f / tot;
#pragma unroll
  for (int e = 0; e < 8; ++e) {
    float v = (sc[q * 128 + g * 8 + e] - mean) * rs;
    attnB[(size_t)bh * 2048 + q * 128 + g * 8 + e] = f2bf(__expf(v - mx) * inv);
  }
}

// ---------------------------------------------------------------------------
// K7: out = attn @ vals : M=16, K=128, N=9216, one wave per (bh, tile)
// ---------------------------------------------------------------------------
__global__ __launch_bounds__(32) void ecm_attnv(
    const unsigned short* __restrict__ attnB, const unsigned short* __restrict__ valsB,
    unsigned short* __restrict__ outB) {
  const int stile = blockIdx.x, bh = blockIdx.y;
  const int lane = threadIdx.x, half = lane >> 4, l15 = lane & 15;
  const int sbase = stile * 16;
  const unsigned short* arow  = attnB + (size_t)bh * 2048 + l15 * 128;
  const unsigned short* bbase = valsB + (size_t)bh * MDIM * HWW + sbase + l15;
  v8f acc = {};
#pragma unroll
  for (int k0 = 0; k0 < MDIM; k0 += 32) {
    Frag16 a, bf;
#pragma unroll
    for (int r = 0; r < 8; ++r) {
      int ka = k0 + half * 16 + 2 * r;
      a.u[r] = *(const unsigned int*)(arow + ka);
      int kb = k0 + half * 16 + 2 * r;
      unsigned int lo = bbase[(size_t)kb * HWW];
      unsigned int hi = bbase[(size_t)(kb + 1) * HWW];
      bf.u[r] = lo | (hi << 16);
    }
    acc = wmma_bf16(a, bf, acc);
  }
  int b = bh >> 2, h = bh & 3;
#pragma unroll
  for (int r = 0; r < 8; ++r) {
    int q = r + 8 * half;
    int ch = h * 16 + q;                    // reshape(b, c, h, w)
    outB[((size_t)(b * CC + ch)) * HWW + sbase + l15] = f2bf(acc[r]);
  }
}

// ---------------------------------------------------------------------------
// K8: y = out_w @ out (1x1 conv) + per-channel sum/sumsq for BN
// grid (576, 8), block 128 = 4 waves (one per M-tile of 64 channels)
// ---------------------------------------------------------------------------
__global__ __launch_bounds__(128) void ecm_outconv(
    const unsigned short* __restrict__ outB, const unsigned short* __restrict__ outwB,
    float* __restrict__ yBuf, float* __restrict__ chanSum, float* __restrict__ chanSumSq) {
  const int stile = blockIdx.x, b = blockIdx.y;
  const int tid = threadIdx.x;
  const int wave = tid >> 5, lane = tid & 31, half = lane >> 4, l15 = lane & 15;
  const int sbase = stile * 16;
  __shared__ unsigned short Bt[16][CC];
  __shared__ float yred[CC][16];
  for (int idx = tid; idx < CC * 16; idx += 128) {
    int c = idx >> 4, p = idx & 15;
    Bt[p][c] = outB[((size_t)(b * CC + c)) * HWW + sbase + p];
  }
  __syncthreads();
  const unsigned short* arow = outwB + (size_t)(wave * 16 + l15) * CC;
  v8f acc = {};
#pragma unroll
  for (int k0 = 0; k0 < CC; k0 += 32) {
    Frag16 a, bf;
#pragma unroll
    for (int r = 0; r < 8; ++r) {
      int ka = k0 + half * 16 + 2 * r;
      a.u[r] = *(const unsigned int*)(arow + ka);
      int kb = k0 + half * 16 + 2 * r;
      bf.u[r] = *(const unsigned int*)(&Bt[l15][kb]);
    }
    acc = wmma_bf16(a, bf, acc);
  }
#pragma unroll
  for (int r = 0; r < 8; ++r) {
    int ch = wave * 16 + r + 8 * half;
    float v = acc[r];
    yBuf[((size_t)(b * CC + ch)) * HWW + sbase + l15] = v;
    yred[ch][l15] = v;
  }
  __syncthreads();
  if (tid < CC) {
    float s = 0.f, ss = 0.f;
#pragma unroll
    for (int p = 0; p < 16; ++p) { float v = yred[tid][p]; s += v; ss = fmaf(v, v, ss); }
    atomicAdd(&chanSum[tid], s);
    atomicAdd(&chanSumSq[tid], ss);
  }
}

// ---------------------------------------------------------------------------
// K9: batch-norm (train-mode batch stats) + affine + ReLU -> f32 output
// ---------------------------------------------------------------------------
__global__ __launch_bounds__(256) void ecm_bnrelu(
    const float* __restrict__ yBuf, const float* __restrict__ chanSum,
    const float* __restrict__ chanSumSq, const float* __restrict__ gamma,
    const float* __restrict__ beta, float* __restrict__ out) {
  size_t idx = (size_t)blockIdx.x * 256 + threadIdx.x;
  int ch = (int)((idx / HWW) & (CC - 1));
  const float invN = 1.f / (float)(BB * HWW);
  float mean = chanSum[ch] * invN;
  float var  = chanSumSq[ch] * invN - mean * mean;
  float v = (yBuf[idx] - mean) * rsqrtf(var + 1e-5f) * gamma[ch] + beta[ch];
  out[idx] = fmaxf(v, 0.f);
}

// ---------------------------------------------------------------------------
extern "C" void kernel_launch(void* const* d_in, const int* in_sizes, int n_in,
                              void* d_out, int out_size, void* d_ws, size_t ws_size,
                              hipStream_t stream) {
  (void)in_sizes; (void)n_in; (void)out_size;
  if (ws_size < WS_NEEDED) return;
  const float* cen   = (const float*)d_in[0];
  const float* sumw  = (const float*)d_in[1];
  const float* qw    = (const float*)d_in[2];
  const float* kw    = (const float*)d_in[3];
  const float* vw    = (const float*)d_in[4];
  const float* outw  = (const float*)d_in[5];
  const float* gamma = (const float*)d_in[6];
  const float* beta  = (const float*)d_in[7];

  char* wsb = (char*)d_ws;
  float*          mixedG = (float*)(wsb + OFF_MIXED);
  unsigned short* kwB    = (unsigned short*)(wsb + OFF_KWB);
  unsigned short* vwB    = (unsigned short*)(wsb + OFF_VWB);
  unsigned short* qwB    = (unsigned short*)(wsb + OFF_QWB);
  unsigned short* outwB  = (unsigned short*)(wsb + OFF_OUTWB);
  unsigned short* keysB  = (unsigned short*)(wsb + OFF_KEYSB);
  unsigned short* valsB  = (unsigned short*)(wsb + OFF_VALSB);
  unsigned short* qryB   = (unsigned short*)(wsb + OFF_QRYB);
  float*          invnk  = (float*)(wsb + OFF_INVNK);
  float*          invnq  = (float*)(wsb + OFF_INVNQ);
  float*          scores = (float*)(wsb + OFF_SCORES);
  unsigned short* attnB  = (unsigned short*)(wsb + OFF_ATTNB);
  unsigned short* outB   = (unsigned short*)(wsb + OFF_OUTB);
  float*          yBuf   = (float*)(wsb + OFF_YBUF);
  float*          chanSum  = (float*)(wsb + OFF_CSUM);
  float*          chanSumSq= (float*)(wsb + OFF_CSSQ);

  ecm_prep<<<1024, 256, 0, stream>>>(sumw, qw, kw, vw, outw, mixedG,
                                     qwB, kwB, vwB, outwB, chanSum, chanSumSq);
  ecm_conv_proj<<<dim3(NTILES, BB, NHD), 256, 0, stream>>>(cen, mixedG, kwB, vwB,
                                                           keysB, valsB);
  ecm_qry<<<dim3(NTILES, BB, NHD), 32, 0, stream>>>(cen, qwB, qryB);
  ecm_rownorm<<<4608, 256, 0, stream>>>(keysB, qryB, invnk, invnq);
  ecm_scores<<<32, 256, 0, stream>>>(qryB, keysB, invnq, invnk, scores);
  ecm_attn<<<32, 256, 0, stream>>>(scores, attnB);
  ecm_attnv<<<dim3(NTILES, 32), 32, 0, stream>>>(attnB, valsB, outB);
  ecm_outconv<<<dim3(NTILES, BB), 128, 0, stream>>>(outB, outwB, yBuf,
                                                    chanSum, chanSumSq);
  ecm_bnrelu<<<18432, 256, 0, stream>>>(yBuf, chanSum, chanSumSq, gamma, beta,
                                        (float*)d_out);
}
